// MambaLayer_52604759442063
// MI455X (gfx1250) — matrix operand order repaired and checked by reading, not compiled, not run on hardware
//
#include <hip/hip_runtime.h>
#include <hip/hip_bf16.h>
#include <stdint.h>

// ---------------------------------------------------------------------------
// Types for CDNA5 WMMA
// ---------------------------------------------------------------------------
typedef __attribute__((ext_vector_type(16))) __bf16 v16bf;
typedef __attribute__((ext_vector_type(8)))  float  v8f;

// ---------------------------------------------------------------------------
// Async global->LDS DMA (verified on this toolchain in round 2/3)
// ---------------------------------------------------------------------------
#if defined(__has_builtin)
# if __has_builtin(__builtin_amdgcn_global_load_async_to_lds_b128)
#  define USE_ASYNC_LDS 1
# endif
# if __has_builtin(__builtin_amdgcn_tensor_load_to_lds) && __has_builtin(__builtin_amdgcn_s_wait_tensorcnt)
#  define USE_TDM 1
# endif
#endif
#ifndef USE_ASYNC_LDS
# define USE_ASYNC_LDS 0
#endif
#ifndef USE_TDM
# define USE_TDM 0
#endif

typedef int v4i_t __attribute__((vector_size(16)));
typedef __attribute__((address_space(1))) v4i_t g_v4i;
typedef __attribute__((address_space(3))) v4i_t l_v4i;

// TDM descriptor group vector types (probe doc: g0=uint32x4, g1=int32x8,
// g2/g3=int32x4, [clang-23: +int32x8], cpol i32)
typedef unsigned int tdm_u4 __attribute__((vector_size(16)));
typedef int          tdm_i8 __attribute__((vector_size(32)));
typedef int          tdm_i4 __attribute__((vector_size(16)));

__device__ __forceinline__ g_v4i* gbl_cast4(const void* p) {
    return (g_v4i*)(uintptr_t)p;
}
__device__ __forceinline__ l_v4i* lds_cast4(void* p) {
    // flat LDS address: low 32 bits are the LDS offset (ISA aperture rule)
    return (l_v4i*)(unsigned int)(uintptr_t)p;
}

__device__ __forceinline__ unsigned short f2bf(float f) {
    unsigned int u = __builtin_bit_cast(unsigned int, f);
    unsigned int r = u + 0x7FFFu + ((u >> 16) & 1u);   // round-to-nearest-even
    return (unsigned short)(r >> 16);
}

__device__ __forceinline__ float sigmoidf_(float x) {
    return 1.0f / (1.0f + __expf(-x));
}

#if USE_TDM
// Issue one TDM 2D tile load: W[n0..n0+63][k0..k0+31] (bf16, row stride K)
// into LDS at lds_off with padded row stride 48 elems (16DW data + 8DW pad).
__device__ __forceinline__ void tdm_load_tile(
    const unsigned short* W, int n0, int k0, int N, int K, unsigned int lds_off)
{
    const unsigned long long ga =
        (unsigned long long)(uintptr_t)(W + (size_t)n0 * K + k0);
    tdm_u4 g0;
    g0[0] = 1u;                                  // count=1 (valid user D#)
    g0[1] = lds_off;                             // lds_addr (bytes)
    g0[2] = (unsigned int)ga;                    // global_addr[31:0]
    g0[3] = (unsigned int)((ga >> 32) & 0x01FFFFFFu) | (2u << 30); // addr[56:32]|type=2
    tdm_i8 g1;
    g1[0] = (int)((1u << 16)      // data_size = 2 bytes
                | (1u << 20)      // pad_enable
                | (3u << 22)      // pad_interval: 16 DWORDs (one 64B tile row)
                | (7u << 25));    // pad_amount: 8 DWORDs (32B -> stride 96B)
    g1[1] = (int)(((unsigned)K & 0xFFFFu) << 16);          // tensor_dim0 lo16
    g1[2] = (int)(((unsigned)K >> 16) | (((unsigned)N & 0xFFFFu) << 16)); // dim0 hi | dim1 lo
    g1[3] = (int)(((unsigned)N >> 16) | (32u << 16));      // dim1 hi | tile_dim0=32
    g1[4] = 64;                                            // tile_dim1=64, tile_dim2=0
    g1[5] = K;                                             // tensor_dim0_stride lo32
    g1[6] = 0;                                             // stride hi16 | dim1_stride lo16
    g1[7] = 0;
    tdm_i4 gz = {0, 0, 0, 0};
#if __clang_major__ >= 23
    tdm_i8 gz8 = {0, 0, 0, 0, 0, 0, 0, 0};
    __builtin_amdgcn_tensor_load_to_lds(g0, g1, gz, gz, gz8, 0);
#else
    __builtin_amdgcn_tensor_load_to_lds(g0, g1, gz, gz, 0);
#endif
}
#endif

// ---------------------------------------------------------------------------
// K0: weight conversion to bf16 + A = -exp(A_log)
// ---------------------------------------------------------------------------
__global__ __launch_bounds__(256) void prep_kernel(
    const float* __restrict__ W_in,   unsigned short* __restrict__ wi_bf,
    const float* __restrict__ W_xp,   unsigned short* __restrict__ wx_bf,
    const float* __restrict__ W_out,  unsigned short* __restrict__ wo_bf,
    const float* __restrict__ A_log,  float* __restrict__ A_neg)
{
    int i = blockIdx.x * 256 + threadIdx.x;
    if (i < 1024 * 256) wi_bf[i] = f2bf(W_in[i]);
    if (i < 48 * 512)   wx_bf[i] = f2bf(W_xp[i]);
    if (i < 256 * 512)  wo_bf[i] = f2bf(W_out[i]);
    if (i < 512 * 16)   A_neg[i] = -__expf(A_log[i]);
}

// ---------------------------------------------------------------------------
// K1: LayerNorm over C=256 per token; one block per (b,l); output bf16
// ---------------------------------------------------------------------------
__global__ __launch_bounds__(256) void ln_kernel(
    const float* __restrict__ x, const float* __restrict__ gamma,
    const float* __restrict__ beta, unsigned short* __restrict__ xn_bf)
{
    const int m = blockIdx.x;
    const int b = m >> 12;
    const int l = m & 4095;
    const int c = threadIdx.x;

    float v = x[((size_t)b * 256 + c) * 4096 + l];

    float s1 = v, s2 = v * v;
#pragma unroll
    for (int mask = 16; mask; mask >>= 1) {
        s1 += __shfl_xor(s1, mask);
        s2 += __shfl_xor(s2, mask);
    }
    __shared__ float r1[8], r2[8];
    const int wid = threadIdx.x >> 5, lane = threadIdx.x & 31;
    if (lane == 0) { r1[wid] = s1; r2[wid] = s2; }
    __syncthreads();
    if (threadIdx.x == 0) {
        float a = 0.f, bb = 0.f;
#pragma unroll
        for (int i = 0; i < 8; ++i) { a += r1[i]; bb += r2[i]; }
        r1[0] = a; r2[0] = bb;
    }
    __syncthreads();
    const float mu  = r1[0] * (1.0f / 256.0f);
    const float var = r2[0] * (1.0f / 256.0f) - mu * mu;
    const float xn  = (v - mu) * __frsqrt_rn(var + 1e-5f) * gamma[c] + beta[c];
    xn_bf[(size_t)m * 256 + c] = f2bf(xn);
}

// ---------------------------------------------------------------------------
// Shared WMMA tile pieces (LDS row stride 48 elems = 96B, 16B aligned).
// ---------------------------------------------------------------------------
__device__ __forceinline__ void gemm_mac(
    const unsigned short* AsBuf, const unsigned short* BsBuf,
    int wave, int half, int lr, v8f* acc)
{
    union { v16bf v; unsigned int w[8]; } fa;
    const unsigned short* arow = &AsBuf[(16 * wave + lr) * 48];
#pragma unroll
    for (int j = 0; j < 8; ++j) {
        const int kb = 16 * half + 2 * j;
        fa.w[j] = *reinterpret_cast<const unsigned int*>(arow + kb);
    }
#pragma unroll
    for (int t = 0; t < 4; ++t) {
        union { v16bf v; unsigned int w[8]; } fb;
        const unsigned short* brow = &BsBuf[(16 * t + lr) * 48];
#pragma unroll
        for (int j = 0; j < 8; ++j)
            fb.w[j] = *reinterpret_cast<const unsigned int*>(brow + 16 * half + 2 * j);
        acc[t] = __builtin_amdgcn_wmma_f32_16x16x32_bf16(
            false, fa.v, false, fb.v, (short)0, acc[t], false, false);
    }
}

template <int MODE>
__device__ __forceinline__ void gemm_store(
    float* Cout, const v8f* acc, int m0, int n0,
    int wave, int half, int lr, int N)
{
#pragma unroll
    for (int t = 0; t < 4; ++t) {
        const int n = n0 + 16 * t + lr;
#pragma unroll
        for (int i = 0; i < 8; ++i) {
            const int m = m0 + 16 * wave + 8 * half + i;
            const float v = acc[t][i];
            if (MODE == 0) {
                if (n < N) Cout[(size_t)m * N + n] = v;
            } else {
                const int b = m >> 12, l = m & 4095;
                Cout[((size_t)((b << 8) + n)) * 4096 + l] = v;
            }
        }
    }
}

// ---------------------------------------------------------------------------
// Pipelined WMMA GEMM (M%64==0, N%64==0, K%32==0). Double-buffered LDS:
//   A tiles via GLOBAL_LOAD_ASYNC_TO_LDS_B128 (ASYNCcnt),
//   W tiles via TDM TENSOR_LOAD_TO_LDS issued by wave 0 (TENSORcnt),
//   completion published to the other waves by the workgroup barrier.
// ---------------------------------------------------------------------------
template <int MODE>
__global__ __launch_bounds__(128) void wmma_gemm_pipe(
    const unsigned short* __restrict__ A, const unsigned short* __restrict__ W,
    float* __restrict__ Cout, int M, int N, int K)
{
    __shared__ __align__(16) unsigned short As[2][64 * 48];
    __shared__ __align__(16) unsigned short Bs[2][64 * 48];

    const int tid  = threadIdx.x;
    const int wave = tid >> 5;
    const int lane = tid & 31;
    const int half = lane >> 4;
    const int lr   = lane & 15;
    const int m0 = blockIdx.x * 64;
    const int n0 = blockIdx.y * 64;

    v8f acc[4];
#pragma unroll
    for (int t = 0; t < 4; ++t) acc[t] = v8f{0.f,0.f,0.f,0.f,0.f,0.f,0.f,0.f};

#if USE_ASYNC_LDS
    auto issue = [&](int k0, int buf) {
        // A tile: 64x32 bf16, 256 b128 transfers spread over 128 threads
#pragma unroll
        for (int i = 0; i < 2; ++i) {
            const int u = tid + 128 * i, row = u >> 2, c4 = u & 3;
            __builtin_amdgcn_global_load_async_to_lds_b128(
                gbl_cast4(A + (size_t)(m0 + row) * K + k0 + c4 * 8),
                lds_cast4(&As[buf][row * 48 + c4 * 8]), 0, 0);
        }
#if USE_TDM
        // W tile: one TDM DMA (wave 0 only; TDM moves the whole tile)
        if (wave == 0)
            tdm_load_tile(W, n0, k0, N, K,
                          (unsigned int)(uintptr_t)&Bs[buf][0]);
#else
#pragma unroll
        for (int i = 0; i < 2; ++i) {
            const int u = tid + 128 * i, row = u >> 2, c4 = u & 3;
            __builtin_amdgcn_global_load_async_to_lds_b128(
                gbl_cast4(W + (size_t)(n0 + row) * K + k0 + c4 * 8),
                lds_cast4(&Bs[buf][row * 48 + c4 * 8]), 0, 0);
        }
#endif
    };
    issue(0, 0);
    const int T = K / 32;
    for (int t = 0; t < T; ++t) {
        const int buf = t & 1;
        if (t + 1 < T) {
            issue((t + 1) * 32, buf ^ 1);
#if USE_TDM
            asm volatile("s_wait_asynccnt 0x2" ::: "memory");
            if (wave == 0) __builtin_amdgcn_s_wait_tensorcnt(1);
#else
            asm volatile("s_wait_asynccnt 0x4" ::: "memory");
#endif
        } else {
            asm volatile("s_wait_asynccnt 0x0" ::: "memory");
#if USE_TDM
            if (wave == 0) __builtin_amdgcn_s_wait_tensorcnt(0);
#endif
        }
        __syncthreads();
        gemm_mac(&As[buf][0], &Bs[buf][0], wave, half, lr, acc);
        __syncthreads();
    }
#else
    for (int k0 = 0; k0 < K; k0 += 32) {
#pragma unroll
        for (int i = 0; i < 2; ++i) {
            const int u = tid + 128 * i, row = u >> 2, c4 = u & 3;
            *reinterpret_cast<uint4*>(&As[0][row * 48 + c4 * 8]) =
                *reinterpret_cast<const uint4*>(A + (size_t)(m0 + row) * K + k0 + c4 * 8);
            *reinterpret_cast<uint4*>(&Bs[0][row * 48 + c4 * 8]) =
                *reinterpret_cast<const uint4*>(W + (size_t)(n0 + row) * K + k0 + c4 * 8);
        }
        __syncthreads();
        gemm_mac(&As[0][0], &Bs[0][0], wave, half, lr, acc);
        __syncthreads();
    }
#endif
    gemm_store<MODE>(Cout, acc, m0, n0, wave, half, lr, N);
}

// ---------------------------------------------------------------------------
// Guarded WMMA GEMM for ragged N (x_proj: N=48): sync staging, zero-fill
// out-of-range W rows, guarded stores.
// ---------------------------------------------------------------------------
__global__ __launch_bounds__(128) void wmma_gemm_edge(
    const unsigned short* __restrict__ A, const unsigned short* __restrict__ W,
    float* __restrict__ Cout, int M, int N, int K)
{
    __shared__ __align__(16) unsigned short As[64 * 48];
    __shared__ __align__(16) unsigned short Bs[64 * 48];

    const int tid  = threadIdx.x;
    const int wave = tid >> 5;
    const int lane = tid & 31;
    const int half = lane >> 4;
    const int lr   = lane & 15;
    const int m0 = blockIdx.x * 64;
    const int n0 = blockIdx.y * 64;

    v8f acc[4];
#pragma unroll
    for (int t = 0; t < 4; ++t) acc[t] = v8f{0.f,0.f,0.f,0.f,0.f,0.f,0.f,0.f};

    for (int k0 = 0; k0 < K; k0 += 32) {
#pragma unroll
        for (int i = 0; i < 2; ++i) {
            const int u = tid + 128 * i, row = u >> 2, c4 = u & 3;
            *reinterpret_cast<uint4*>(&As[row * 48 + c4 * 8]) =
                *reinterpret_cast<const uint4*>(A + (size_t)(m0 + row) * K + k0 + c4 * 8);
            uint4 bv = make_uint4(0u, 0u, 0u, 0u);
            const int nrow = n0 + row;
            if (nrow < N)
                bv = *reinterpret_cast<const uint4*>(W + (size_t)nrow * K + k0 + c4 * 8);
            *reinterpret_cast<uint4*>(&Bs[row * 48 + c4 * 8]) = bv;
        }
        __syncthreads();
        gemm_mac(&As[0], &Bs[0], wave, half, lr, acc);
        __syncthreads();
    }
    gemm_store<0>(Cout, acc, m0, n0, wave, half, lr, N);
}

// ---------------------------------------------------------------------------
// K3: depthwise causal conv(width 4) + SiLU.  xp = xz[:, :512]
// ---------------------------------------------------------------------------
__global__ __launch_bounds__(256) void conv_kernel(
    const float* __restrict__ xz, const float* __restrict__ cw,
    const float* __restrict__ cb, float* __restrict__ xc,
    unsigned short* __restrict__ xc_bf)
{
    const int idx = blockIdx.x * 256 + threadIdx.x;
    const int m = idx >> 9, d = idx & 511;
    const int b = m >> 12, l = m & 4095;
    float acc = cb[d];
#pragma unroll
    for (int t = 0; t < 4; ++t) {
        const int ll = l - 3 + t;
        if (ll >= 0) {
            const size_t mm = ((size_t)b << 12) + ll;
            acc += cw[d * 4 + t] * xz[mm * 1024 + d];
        }
    }
    const float s = acc * sigmoidf_(acc);
    xc[idx]    = s;
    xc_bf[idx] = f2bf(s);
}

// ---------------------------------------------------------------------------
// K4: delta = softplus(dt @ W_dt^T + b_dt); pack (delta, xc) as float2
// ---------------------------------------------------------------------------
__global__ __launch_bounds__(256) void delta_kernel(
    const float* __restrict__ dbl, const float* __restrict__ W_dt,
    const float* __restrict__ b_dt, const float* __restrict__ xc,
    float2* __restrict__ dx)
{
    const int idx = blockIdx.x * 256 + threadIdx.x;
    const int m = idx >> 9, d = idx & 511;
    float a = b_dt[d];
#pragma unroll
    for (int r = 0; r < 16; ++r)
        a += dbl[(size_t)m * 48 + r] * W_dt[d * 16 + r];
    const float dlt = (a > 20.0f) ? a : log1pf(__expf(a));
    dx[idx] = make_float2(dlt, xc[idx]);
}

// ---------------------------------------------------------------------------
// K5: selective scan. One (b,d) channel per 16-lane half-wave; lane = state n.
// ---------------------------------------------------------------------------
__global__ __launch_bounds__(256) void scan_kernel(
    const float2* __restrict__ dx,   const float* __restrict__ dbl,
    const float* __restrict__ xz,    const float* __restrict__ A_neg,
    const float* __restrict__ Dv,    unsigned short* __restrict__ y_bf)
{
    const int tid = threadIdx.x;
    const int n   = tid & 15;
    const int ch  = blockIdx.x * 16 + (tid >> 4);
    const int b   = ch >> 9, d = ch & 511;

    const float An = A_neg[d * 16 + n];
    const float Dd = Dv[d];
    float h = 0.0f;
    const size_t base = (size_t)b << 12;

    for (int l = 0; l < 4096; ++l) {
        const size_t m = base + l;
        __builtin_prefetch(&dbl[(m + 8) * 48], 0, 1);   // global_prefetch_b8
        const float2 dxv = dx[m * 512 + d];
        const float dv = dxv.x, xv = dxv.y;
        const float Bn = dbl[m * 48 + 16 + n];
        const float Cn = dbl[m * 48 + 32 + n];
        h = __expf(dv * An) * h + (dv * xv) * Bn;
        float p = h * Cn;
        p += __shfl_xor(p, 1);
        p += __shfl_xor(p, 2);
        p += __shfl_xor(p, 4);
        p += __shfl_xor(p, 8);
        if (n == 0) {
            const float zv = xz[m * 1024 + 512 + d];
            const float yv = (p + xv * Dd) * (zv * sigmoidf_(zv));
            y_bf[m * 512 + d] = f2bf(yv);
        }
    }
}

// ---------------------------------------------------------------------------
// Host launcher
// ---------------------------------------------------------------------------
extern "C" void kernel_launch(void* const* d_in, const int* in_sizes, int n_in,
                              void* d_out, int out_size, void* d_ws, size_t ws_size,
                              hipStream_t stream) {
    (void)in_sizes; (void)n_in; (void)out_size; (void)ws_size;

    const float* x       = (const float*)d_in[0];
    const float* ln_g    = (const float*)d_in[1];
    const float* ln_b    = (const float*)d_in[2];
    const float* W_in    = (const float*)d_in[3];
    const float* conv_w  = (const float*)d_in[4];
    const float* conv_b  = (const float*)d_in[5];
    const float* W_xproj = (const float*)d_in[6];
    const float* W_dt    = (const float*)d_in[7];
    const float* b_dt    = (const float*)d_in[8];
    const float* A_log   = (const float*)d_in[9];
    const float* Dvec    = (const float*)d_in[10];
    const float* W_out   = (const float*)d_in[11];
    float* out = (float*)d_out;

    char* ws = (char*)d_ws;
    size_t off = 0;
    auto alloc = [&](size_t bytes) {
        char* p = ws + off;
        off += (bytes + 255) & ~(size_t)255;
        return p;
    };
    unsigned short* wi_bf  = (unsigned short*)alloc(1024 * 256 * 2);
    unsigned short* wx_bf  = (unsigned short*)alloc(48 * 512 * 2);
    unsigned short* wo_bf  = (unsigned short*)alloc(256 * 512 * 2);
    float*          a_neg  = (float*)         alloc(512 * 16 * 4);
    unsigned short* xn_bf  = (unsigned short*)alloc((size_t)8192 * 256 * 2);
    float*          xz     = (float*)         alloc((size_t)8192 * 1024 * 4);
    float*          xc     = (float*)         alloc((size_t)8192 * 512 * 4);
    unsigned short* xc_bf  = (unsigned short*)alloc((size_t)8192 * 512 * 2);
    float*          dbl    = (float*)         alloc((size_t)8192 * 48 * 4);
    float2*         dx     = (float2*)        alloc((size_t)8192 * 512 * 8);
    unsigned short* y_bf   = (unsigned short*)alloc((size_t)8192 * 512 * 2);

    prep_kernel<<<(1024 * 256 + 255) / 256, 256, 0, stream>>>(
        W_in, wi_bf, W_xproj, wx_bf, W_out, wo_bf, A_log, a_neg);

    ln_kernel<<<8192, 256, 0, stream>>>(x, ln_g, ln_b, xn_bf);

    // xz = xn @ W_in^T (8192x1024, K=256) — async A + TDM W, double-buffered
    wmma_gemm_pipe<0><<<dim3(8192 / 64, 1024 / 64), 128, 0, stream>>>(
        xn_bf, wi_bf, xz, 8192, 1024, 256);

    conv_kernel<<<(8192 * 512) / 256, 256, 0, stream>>>(
        xz, conv_w, conv_b, xc, xc_bf);

    // dbl = xc @ W_xproj^T (8192x48, K=512) — guarded WMMA
    wmma_gemm_edge<<<dim3(8192 / 64, 1), 128, 0, stream>>>(
        xc_bf, wx_bf, dbl, 8192, 48, 512);

    delta_kernel<<<(8192 * 512) / 256, 256, 0, stream>>>(
        dbl, W_dt, b_dt, xc, dx);

    scan_kernel<<<64, 256, 0, stream>>>(
        dx, dbl, xz, a_neg, Dvec, y_bf);

    // out = y @ W_out^T (8192x256, K=512), transposed (B,C,L) store
    wmma_gemm_pipe<1><<<dim3(8192 / 64, 256 / 64), 128, 0, stream>>>(
        y_bf, wo_bf, out, 8192, 256, 512);
}
